// RegionProposalNetwork_30039001268644
// MI455X (gfx1250) — matrix operand order repaired and checked by reading, not compiled, not run on hardware
//
#include <hip/hip_runtime.h>
#include <hip/hip_bf16.h>
#include <math.h>
#include <stdint.h>

// ---------------------------------------------------------------------------
// RPN forward for MI455X (gfx1250, wave32, WMMA + async-to-LDS staging).
// 3x3 conv 512->512 as implicit GEMM (M=16384,K=4608,N=512) on
// v_wmma_f32_16x16x32_f16 with double-buffered GLOBAL_LOAD_ASYNC_TO_LDS_B128;
// 1x1 convs fused as one GEMM (N padded to 64).
// ---------------------------------------------------------------------------

typedef __attribute__((ext_vector_type(16))) _Float16 v16h;
typedef __attribute__((ext_vector_type(8)))  _Float16 v8h;
typedef __attribute__((ext_vector_type(8)))  float    v8f;

#define NIMG 4
#define HH 64
#define WW 64
#define CIN 512
#define CMID 512
#define NA 9
#define KANCH (HH*WW*NA)          // 36864
#define MTOT (NIMG*HH*WW)         // 16384
#define NPOST 2000
#define PADW 66
#define NEG_INF (-__builtin_inff())

// d_out float element offsets (return-order concatenation)
#define OFF_LOCS   0u
#define OFF_SCORES 589824u
#define OFF_ROIS   884736u
#define OFF_IDX    916736u
#define OFF_ANCHOR 924736u

// ---------------------------------------------------------------------------
// CDNA5 async-to-LDS helpers (ASYNCcnt path, CDNA5 ISA)
// ---------------------------------------------------------------------------
__device__ __forceinline__ void async_load_b128(uint32_t lds_byte, const void* gaddr) {
  // LDS[vdst + lane] = MEM[vaddr + lane]; tracked by ASYNCcnt
  asm volatile("global_load_async_to_lds_b128 %0, %1, off"
               :: "v"(lds_byte), "v"((unsigned long long)(uintptr_t)gaddr)
               : "memory");
}

__device__ __forceinline__ void wait_async0() {
#if __has_builtin(__builtin_amdgcn_s_wait_asynccnt)
  __builtin_amdgcn_s_wait_asynccnt(0);
#else
  asm volatile("s_wait_asynccnt 0x0" ::: "memory");
#endif
}

// ---------------------------------------------------------------------------
// Prep kernels
// ---------------------------------------------------------------------------

// x: NCHW f32 -> zero-padded NHWC f16 [4][66][66][512]
__global__ void cvt_x_pad_kernel(const float* __restrict__ x, _Float16* __restrict__ xhp) {
  const int total = NIMG * PADW * PADW * CIN;
  for (int o = blockIdx.x * blockDim.x + threadIdx.x; o < total;
       o += gridDim.x * blockDim.x) {
    int c = o & 511;
    int t = o >> 9;
    int xx = t % PADW; t /= PADW;
    int yy = t % PADW;
    int n  = t / PADW;
    float v = 0.0f;
    if (yy >= 1 && yy <= 64 && xx >= 1 && xx <= 64)
      v = x[(((size_t)n * CIN + c) << 12) + ((yy - 1) << 6) + (xx - 1)];
    xhp[o] = (_Float16)v;
  }
}

// W1: OIHW [512][512][3][3] f32 -> Wt [p=ky*3+kx][cout][cin] f16
__global__ void cvt_w1_kernel(const float* __restrict__ W1, _Float16* __restrict__ Wt) {
  const int total = 9 * CMID * CIN;
  for (int o = blockIdx.x * blockDim.x + threadIdx.x; o < total;
       o += gridDim.x * blockDim.x) {
    int ci = o & 511;
    int co = (o >> 9) & 511;
    int p  = o >> 18;
    Wt[o] = (_Float16)W1[((size_t)co * CIN + ci) * 9 + p];
  }
}

// Pack Ws(18ch)+Wl(36ch) into Wsl[64][512] f16 + bias[64] f32
__global__ void build_wsl_kernel(const float* __restrict__ Ws, const float* __restrict__ bs,
                                 const float* __restrict__ Wl, const float* __restrict__ bl,
                                 _Float16* __restrict__ Wsl, float* __restrict__ biasC) {
  const int total = 64 * 512;
  for (int o = blockIdx.x * blockDim.x + threadIdx.x; o < total;
       o += gridDim.x * blockDim.x) {
    int ci = o & 511;
    int j  = o >> 9;
    float v = 0.0f;
    if (j < 18)      v = Ws[j * 512 + ci];
    else if (j < 54) v = Wl[(j - 18) * 512 + ci];
    Wsl[o] = (_Float16)v;
    if (ci == 0) {
      float b = 0.0f;
      if (j < 18)      b = bs[j];
      else if (j < 54) b = bl[j - 18];
      biasC[j] = b;
    }
  }
}

// anchors -> d_out[OFF_ANCHOR..]
__global__ void anchors_kernel(float* __restrict__ anc) {
  int k = blockIdx.x * blockDim.x + threadIdx.x;
  if (k >= KANCH) return;
  int a = k % NA;
  int pos = k / NA;
  int xx = pos & 63, yy = pos >> 6;
  const float RA[3] = {0.5f, 1.0f, 2.0f};
  const float SC[3] = {8.0f, 16.0f, 32.0f};
  int ri = a / 3, si = a % 3;
  float ah = 16.0f * SC[si] * sqrtf(RA[ri]);
  float aw = 16.0f * SC[si] * sqrtf(1.0f / RA[ri]);
  float cy = yy * 16.0f + 8.0f;
  float cx = xx * 16.0f + 8.0f;
  anc[(size_t)k * 4 + 0] = cy - 0.5f * ah;
  anc[(size_t)k * 4 + 1] = cx - 0.5f * aw;
  anc[(size_t)k * 4 + 2] = cy + 0.5f * ah;
  anc[(size_t)k * 4 + 3] = cx + 0.5f * aw;
}

// ---------------------------------------------------------------------------
// WMMA fragment loaders (wave32 layouts, ISA 7.12.2)
// ---------------------------------------------------------------------------
__device__ __forceinline__ v16h load_afrag(const _Float16* As, int row, int lane) {
  int klo = (lane >> 4) << 3;
  const v8h lo = *(const v8h*)(As + row * 32 + klo);
  const v8h hi = *(const v8h*)(As + row * 32 + klo + 16);
  return __builtin_shufflevector(lo, hi, 0,1,2,3,4,5,6,7,8,9,10,11,12,13,14,15);
}

__device__ __forceinline__ v16h load_bfrag(const _Float16* Bs, int col, int lane) {
  int klo = (lane >> 4) << 3;
  const v8h lo = *(const v8h*)(Bs + col * 32 + klo);
  const v8h hi = *(const v8h*)(Bs + col * 32 + klo + 16);
  return __builtin_shufflevector(lo, hi, 0,1,2,3,4,5,6,7,8,9,10,11,12,13,14,15);
}

// ---------------------------------------------------------------------------
// Conv 3x3 implicit GEMM: M=16384, K=9*512, N=512.
// Block tile 128x128, 8 waves (4M x 2N), wave tile 32x64 (acc[2][4]).
// K-step 32, double-buffered async-to-LDS staging.
// ---------------------------------------------------------------------------
__global__ __launch_bounds__(256) void
conv3_gemm_kernel(const _Float16* __restrict__ xhp, const _Float16* __restrict__ Wt,
                  const float* __restrict__ b1, _Float16* __restrict__ hout) {
  __shared__ alignas(16) _Float16 As[2][128 * 32];
  __shared__ alignas(16) _Float16 Bs[2][128 * 32];
  const int tid  = threadIdx.x;
  const int lane = tid & 31;
  const int wave = tid >> 5;
  const int waveM = wave & 3;   // 4 waves x 32 rows
  const int waveN = wave >> 2;  // 2 waves x 64 cols
  const int M0 = blockIdx.x * 128;
  const int N0 = blockIdx.y * 128;

  // Per-thread staging geometry: 2 A-chunks + 2 B-chunks of 16B per K-step.
  const int seg = tid & 3;
  int rr0 = tid >> 2;           // 0..63
  int rr1 = rr0 + 64;           // 64..127
  // A rows -> padded NHWC position base
  int rowPos[2];
#pragma unroll
  for (int it = 0; it < 2; ++it) {
    int m = M0 + (it ? rr1 : rr0);
    int n = m >> 12, rem = m & 4095, yy = rem >> 6, xx = rem & 63;
    rowPos[it] = (n * PADW + yy) * PADW + xx;   // (+ky*66+kx) later; pad=1 folded in
  }
  uint32_t aLds[2], bLds[2];
  aLds[0] = (uint32_t)(uintptr_t)&As[0][rr0 * 32 + seg * 8];
  aLds[1] = (uint32_t)(uintptr_t)&As[0][rr1 * 32 + seg * 8];
  bLds[0] = (uint32_t)(uintptr_t)&Bs[0][rr0 * 32 + seg * 8];
  bLds[1] = (uint32_t)(uintptr_t)&Bs[0][rr1 * 32 + seg * 8];
  const uint32_t bufStride = 128 * 32 * sizeof(_Float16);  // 8192

  v8f acc[2][4] = {};

  auto stage = [&](int s, int b) {
    int p  = s >> 4;
    int c0 = (s & 15) << 5;
    int ky = p / 3, kx = p % 3;
    int koff = (ky * PADW + kx) * CIN + c0 + seg * 8;
#pragma unroll
    for (int it = 0; it < 2; ++it)
      async_load_b128(aLds[it] + (uint32_t)b * bufStride,
                      xhp + (size_t)rowPos[it] * CIN + koff);
    int wbase = ((p * 512 + N0) << 9) + c0 + seg * 8;
#pragma unroll
    for (int it = 0; it < 2; ++it)
      async_load_b128(bLds[it] + (uint32_t)b * bufStride,
                      Wt + (size_t)wbase + ((it ? rr1 : rr0) << 9));
  };

  stage(0, 0);
  wait_async0();
  __syncthreads();

  const int NSTEP = 9 * (CIN / 32);  // 144
  for (int s = 0; s < NSTEP; ++s) {
    int b = s & 1;
    if (s + 1 < NSTEP) stage(s + 1, b ^ 1);

    v16h af[2], bf[4];
#pragma unroll
    for (int mt = 0; mt < 2; ++mt)
      af[mt] = load_afrag(As[b], waveM * 32 + mt * 16 + (lane & 15), lane);
#pragma unroll
    for (int nt = 0; nt < 4; ++nt)
      bf[nt] = load_bfrag(Bs[b], waveN * 64 + nt * 16 + (lane & 15), lane);
#pragma unroll
    for (int mt = 0; mt < 2; ++mt)
#pragma unroll
      for (int nt = 0; nt < 4; ++nt)
        acc[mt][nt] = __builtin_amdgcn_wmma_f32_16x16x32_f16(
            false, af[mt], false, bf[nt], (short)0, acc[mt][nt], false, false);

    wait_async0();
    __syncthreads();
  }

  // Epilogue: bias + ReLU -> f16 h[M][512]
#pragma unroll
  for (int mt = 0; mt < 2; ++mt)
#pragma unroll
    for (int nt = 0; nt < 4; ++nt) {
      int col = N0 + waveN * 64 + nt * 16 + (lane & 15);
      float bias = b1[col];
      int rbase = waveM * 32 + mt * 16 + ((lane >> 4) << 3);
#pragma unroll
      for (int r = 0; r < 8; ++r) {
        float v = acc[mt][nt][r] + bias;
        v = v > 0.0f ? v : 0.0f;
        hout[((size_t)(M0 + rbase + r) << 9) + col] = (_Float16)v;
      }
    }
}

// ---------------------------------------------------------------------------
// Dense GEMM for fused 1x1 convs: M=16384, K=512, N=64.
// Block tile 128x64, waves 4M x 2N (32x32 each), double-buffered async staging.
// ---------------------------------------------------------------------------
__global__ __launch_bounds__(256) void
dense_gemm_kernel(const _Float16* __restrict__ h, const _Float16* __restrict__ Wsl,
                  const float* __restrict__ biasC, float* __restrict__ out2) {
  __shared__ alignas(16) _Float16 As[2][128 * 32];
  __shared__ alignas(16) _Float16 Bs[2][64 * 32];
  const int tid  = threadIdx.x;
  const int lane = tid & 31;
  const int wave = tid >> 5;
  const int waveM = wave & 3;
  const int waveN = wave >> 2;
  const int M0 = blockIdx.x * 128;

  const int seg = tid & 3;
  int rr0 = tid >> 2;          // 0..63 (A rows 0..63 and B rows)
  int rr1 = rr0 + 64;          // A rows 64..127
  uint32_t aLds[2], bLds;
  aLds[0] = (uint32_t)(uintptr_t)&As[0][rr0 * 32 + seg * 8];
  aLds[1] = (uint32_t)(uintptr_t)&As[0][rr1 * 32 + seg * 8];
  bLds    = (uint32_t)(uintptr_t)&Bs[0][rr0 * 32 + seg * 8];
  const uint32_t aStride = 128 * 32 * sizeof(_Float16);
  const uint32_t bStride = 64 * 32 * sizeof(_Float16);

  v8f acc[2][2] = {};

  auto stage = [&](int s, int b) {
    int c0 = (s << 5) + seg * 8;
#pragma unroll
    for (int it = 0; it < 2; ++it)
      async_load_b128(aLds[it] + (uint32_t)b * aStride,
                      h + ((size_t)(M0 + (it ? rr1 : rr0)) << 9) + c0);
    async_load_b128(bLds + (uint32_t)b * bStride, Wsl + ((size_t)rr0 << 9) + c0);
  };

  stage(0, 0);
  wait_async0();
  __syncthreads();

  const int NSTEP = CIN / 32;  // 16
  for (int s = 0; s < NSTEP; ++s) {
    int b = s & 1;
    if (s + 1 < NSTEP) stage(s + 1, b ^ 1);

    v16h af[2], bf[2];
#pragma unroll
    for (int mt = 0; mt < 2; ++mt)
      af[mt] = load_afrag(As[b], waveM * 32 + mt * 16 + (lane & 15), lane);
#pragma unroll
    for (int nt = 0; nt < 2; ++nt)
      bf[nt] = load_bfrag(Bs[b], waveN * 32 + nt * 16 + (lane & 15), lane);
#pragma unroll
    for (int mt = 0; mt < 2; ++mt)
#pragma unroll
      for (int nt = 0; nt < 2; ++nt)
        acc[mt][nt] = __builtin_amdgcn_wmma_f32_16x16x32_f16(
            false, af[mt], false, bf[nt], (short)0, acc[mt][nt], false, false);

    wait_async0();
    __syncthreads();
  }

#pragma unroll
  for (int mt = 0; mt < 2; ++mt)
#pragma unroll
    for (int nt = 0; nt < 2; ++nt) {
      int col = waveN * 32 + nt * 16 + (lane & 15);
      float bias = biasC[col];
      int rbase = waveM * 32 + mt * 16 + ((lane >> 4) << 3);
#pragma unroll
      for (int r = 0; r < 8; ++r)
        out2[((size_t)(M0 + rbase + r) << 6) + col] = acc[mt][nt][r] + bias;
    }
}

// ---------------------------------------------------------------------------
// Scatter GEMM output into reference layouts + bbox decode + size filter.
// ---------------------------------------------------------------------------
__global__ void decode_kernel(const float* __restrict__ out2,
                              float* __restrict__ rpn_locs, float* __restrict__ rpn_scores,
                              float* __restrict__ roi, float* __restrict__ fg,
                              const int* __restrict__ img_h_p, const int* __restrict__ img_w_p) {
  int idx = blockIdx.x * blockDim.x + threadIdx.x;
  if (idx >= MTOT * NA) return;
  int m = idx / NA, a = idx - m * NA;
  int n = m >> 12, rem = m & 4095, yy = rem >> 6, xx = rem & 63;
  int kidx = (yy * 64 + xx) * NA + a;
  const float* o = out2 + ((size_t)m << 6);

  float s0 = o[a * 2 + 0], s1 = o[a * 2 + 1];
  size_t sb = ((size_t)n * KANCH + kidx) * 2;
  rpn_scores[sb] = s0;
  rpn_scores[sb + 1] = s1;

  float dy = o[18 + a * 4 + 0], dx = o[18 + a * 4 + 1];
  float dh = o[18 + a * 4 + 2], dw = o[18 + a * 4 + 3];
  size_t lb = ((size_t)n * KANCH + kidx) * 4;
  rpn_locs[lb + 0] = dy; rpn_locs[lb + 1] = dx;
  rpn_locs[lb + 2] = dh; rpn_locs[lb + 3] = dw;

  const float RA[3] = {0.5f, 1.0f, 2.0f};
  const float SC[3] = {8.0f, 16.0f, 32.0f};
  int ri = a / 3, si = a % 3;
  float ah = 16.0f * SC[si] * sqrtf(RA[ri]);
  float aw = 16.0f * SC[si] * sqrtf(1.0f / RA[ri]);
  float acy = yy * 16.0f + 8.0f;
  float acx = xx * 16.0f + 8.0f;

  float cy = dy * ah + acy, cx = dx * aw + acx;
  float bh = expf(dh) * ah, bw = expf(dw) * aw;
  float Hc = (float)img_h_p[0], Wc = (float)img_w_p[0];
  float y1 = fminf(fmaxf(cy - 0.5f * bh, 0.0f), Hc);
  float x1 = fminf(fmaxf(cx - 0.5f * bw, 0.0f), Wc);
  float y2 = fminf(fmaxf(cy + 0.5f * bh, 0.0f), Hc);
  float x2 = fminf(fmaxf(cx + 0.5f * bw, 0.0f), Wc);
  roi[lb + 0] = y1; roi[lb + 1] = x1; roi[lb + 2] = y2; roi[lb + 3] = x2;

  bool valid = ((y2 - y1) >= 16.0f) && ((x2 - x1) >= 16.0f);
  fg[(size_t)n * KANCH + kidx] = valid ? s1 : NEG_INF;
}

// ---------------------------------------------------------------------------
// Greedy NMS, one 1024-thread workgroup per image.
// ---------------------------------------------------------------------------
__global__ __launch_bounds__(1024) void
nms_kernel(const float* __restrict__ roi, float* __restrict__ fg,
           float* __restrict__ rois_out, float* __restrict__ idx_out) {
  const int img = blockIdx.x;
  const int tid = threadIdx.x;
  const float* boxes = roi + (size_t)img * KANCH * 4;
  float* sc = fg + (size_t)img * KANCH;

  __shared__ float rs[1024];
  __shared__ int   rj[1024];
  __shared__ float bb[5];

  if (tid < NPOST) idx_out[img * NPOST + tid] = (float)img;

  for (int i = 0; i < NPOST; ++i) {
    float best = NEG_INF;
    int bi = -1;
    for (int k = tid; k < KANCH; k += 1024) {
      float v = sc[k];
      if (v > best) { best = v; bi = k; }
    }
    rs[tid] = best;
    rj[tid] = bi;
    __syncthreads();
    for (int s = 512; s > 0; s >>= 1) {
      if (tid < s && rs[tid + s] > rs[tid]) {
        rs[tid] = rs[tid + s];
        rj[tid] = rj[tid + s];
      }
      __syncthreads();
    }
    if (tid == 0) {
      int j = rj[0];
      bool valid = (j >= 0) && (rs[0] > NEG_INF);
      float y1 = 0.f, x1 = 0.f, y2 = 0.f, x2 = 0.f;
      if (valid) {
        y1 = boxes[j * 4 + 0]; x1 = boxes[j * 4 + 1];
        y2 = boxes[j * 4 + 2]; x2 = boxes[j * 4 + 3];
        sc[j] = NEG_INF;
      }
      size_t ob = ((size_t)img * NPOST + i) * 4;
      rois_out[ob + 0] = y1; rois_out[ob + 1] = x1;
      rois_out[ob + 2] = y2; rois_out[ob + 3] = x2;
      bb[0] = y1; bb[1] = x1; bb[2] = y2; bb[3] = x2;
      bb[4] = valid ? (y2 - y1) * (x2 - x1) : -1.0f;
    }
    __syncthreads();
    float by1 = bb[0], bx1 = bb[1], by2 = bb[2], bx2 = bb[3], ba = bb[4];
    if (ba >= 0.0f) {
      for (int k = tid; k < KANCH; k += 1024) {
        float v = sc[k];
        if (v == NEG_INF) continue;
        float cy1 = boxes[k * 4 + 0], cx1 = boxes[k * 4 + 1];
        float cy2 = boxes[k * 4 + 2], cx2 = boxes[k * 4 + 3];
        float yy1 = fmaxf(by1, cy1), xx1 = fmaxf(bx1, cx1);
        float yy2 = fminf(by2, cy2), xx2 = fminf(bx2, cx2);
        float ih = fmaxf(yy2 - yy1, 0.0f), iw = fmaxf(xx2 - xx1, 0.0f);
        float inter = ih * iw;
        float ca = (cy2 - cy1) * (cx2 - cx1);
        float iou = inter / (ba + ca - inter + 1e-9f);
        if (iou > 0.7f) sc[k] = NEG_INF;
      }
    }
    __syncthreads();
  }
}

// ---------------------------------------------------------------------------
// Host launcher
// ---------------------------------------------------------------------------
extern "C" void kernel_launch(void* const* d_in, const int* in_sizes, int n_in,
                              void* d_out, int out_size, void* d_ws, size_t ws_size,
                              hipStream_t stream) {
  const float* x  = (const float*)d_in[0];
  const float* W1 = (const float*)d_in[1];
  const float* b1 = (const float*)d_in[2];
  const float* Ws = (const float*)d_in[3];
  const float* bs = (const float*)d_in[4];
  const float* Wl = (const float*)d_in[5];
  const float* bl = (const float*)d_in[6];
  const int* img_h = (const int*)d_in[7];
  const int* img_w = (const int*)d_in[8];
  float* out = (float*)d_out;

  // Workspace layout (bytes)
  char* ws = (char*)d_ws;
  _Float16* xhp   = (_Float16*)(ws + 0);                 // 17,842,176
  _Float16* Wt    = (_Float16*)(ws + 17842176);          //  4,718,592
  _Float16* Wsl   = (_Float16*)(ws + 22560768);          //     65,536
  float*    biasC = (float*)   (ws + 22626304);          //        512
  _Float16* hbuf  = (_Float16*)(ws + 22626816);          // 16,777,216
  float*    out2  = (float*)   (ws + 39404032);          //  4,194,304
  float*    roiB  = (float*)   (ws + 43598336);          //  2,359,296
  float*    fgB   = (float*)   (ws + 45957632);          //    589,824

  // 1) layout/dtype prep
  cvt_x_pad_kernel<<<4096, 256, 0, stream>>>(x, xhp);
  cvt_w1_kernel<<<2048, 256, 0, stream>>>(W1, Wt);
  build_wsl_kernel<<<128, 256, 0, stream>>>(Ws, bs, Wl, bl, Wsl, biasC);
  anchors_kernel<<<(KANCH + 255) / 256, 256, 0, stream>>>(out + OFF_ANCHOR);

  // 2) 3x3 conv + ReLU via WMMA implicit GEMM: grid = (M/128, N/128)
  conv3_gemm_kernel<<<dim3(MTOT / 128, CMID / 128), 256, 0, stream>>>(xhp, Wt, b1, hbuf);

  // 3) fused 1x1 convs via WMMA GEMM (N padded to 64)
  dense_gemm_kernel<<<dim3(MTOT / 128, 1), 256, 0, stream>>>(hbuf, Wsl, biasC, out2);

  // 4) scatter to reference layout + bbox decode + size filter
  decode_kernel<<<(MTOT * NA + 255) / 256, 256, 0, stream>>>(
      out2, out + OFF_LOCS, out + OFF_SCORES, roiB, fgB, img_h, img_w);

  // 5) greedy NMS per image -> rois + indices
  nms_kernel<<<NIMG, 1024, 0, stream>>>(roiB, fgB, out + OFF_ROIS, out + OFF_IDX);
}
